// DeepFace_23167053595205
// MI455X (gfx1250) — matrix operand order repaired and checked by reading, not compiled, not run on hardware
//
#include <hip/hip_runtime.h>
#include <hip/hip_bf16.h>

typedef __attribute__((ext_vector_type(16))) __bf16 v16bf;
typedef __attribute__((ext_vector_type(8)))  float  v8f;

// ---------------------------------------------------------------------------
// CDNA5 WMMA bf16 16x16x32 fragment layouts (ISA 7.12.2), re-derived as
// lane-independent bit ops for element e of the v16bf fragment:
//   A (16x32): row M = lane&15 ; K offset = (e&7) + 8*(lane>=16) + 16*(e>>3)
//   B (32x16): col N = lane&15 ; K offset = (e&7) + 8*(lane>=16) + 16*(e>>3)
//   C/D f32  : VGPR r -> M = r + 8*(lane>=16), N = lane&15

__device__ __forceinline__ float relu_f(float v) { return v > 0.f ? v : 0.f; }

// ---------------------------------------------------------------------------
// Kernel 1: conv1 (1->32, 11x11) + ReLU + fractional-max-pool stand-in,
// fused so the 165MB conv1 intermediate never reaches HBM.
__global__ void conv1_fmp_kernel(const float* __restrict__ x,
                                 const float* __restrict__ w1,
                                 float* __restrict__ out) {
    __shared__ float w1s[121];
    const int c = blockIdx.y, b = blockIdx.z;
    if (threadIdx.x < 121) w1s[threadIdx.x] = w1[c * 121 + threadIdx.x];
    __syncthreads();
    const int idx = blockIdx.x * blockDim.x + threadIdx.x;
    if (idx >= 71 * 71) return;
    const int py = idx / 71, px = idx % 71;
    const float* xb = x + (size_t)b * 152 * 152;
    float best = -1e30f;
    for (int wy = 0; wy < 3; ++wy) {
        const int y = 2 * py + wy;
        if (y >= 142) break;                      // -inf pad never wins
        for (int wx = 0; wx < 3; ++wx) {
            const int xx = 2 * px + wx;
            if (xx >= 142) break;
            float s = 0.f;
            #pragma unroll
            for (int ky = 0; ky < 11; ++ky) {
                const float* row = xb + (size_t)(y + ky) * 152 + xx;
                #pragma unroll
                for (int kx = 0; kx < 11; ++kx)
                    s = fmaf(row[kx], w1s[ky * 11 + kx], s);
            }
            best = fmaxf(best, relu_f(s));
        }
    }
    out[(((size_t)b * 32 + c) * 71 + py) * 71 + px] = best;
}

// ---------------------------------------------------------------------------
// Kernel 2: conv2 (32->16, 9x9) implicit GEMM. K reordered: the 81 spatial
// taps are the outer loop, the 32 input channels are one exact K=32 WMMA
// step.  All per-element strides (c*5041, c*81) are loop-invariant -> no
// div/mod in the hot loop.  One wave per 16x16 D tile, M = 64*63*63.
__global__ void conv2_wmma_kernel(const float* __restrict__ pool,
                                  const float* __restrict__ w2,
                                  const float* __restrict__ b2,
                                  float* __restrict__ out) {
    const int lane = threadIdx.x & 31;
    const int LH   = (lane >= 16) ? 1 : 0;
    const int wave = blockIdx.x * (blockDim.x >> 5) + (threadIdx.x >> 5);
    const int MT = 254016 / 16;                    // 15876 tiles (exact)
    if (wave >= MT) return;                        // wave-uniform
    const int m0 = wave * 16;
    const int ma = m0 + (lane & 15);               // A row (fixed per lane)
    const int ba = ma / 3969, ra = ma % 3969;
    const int ya = ra / 63,   xa = ra % 63;
    const float* abase = pool + ((size_t)ba * 32) * 5041 + (size_t)ya * 71 + xa;
    const int n = lane & 15;
    const float* bbase = w2 + (size_t)n * 2592;    // + c*81 + kk
    v8f acc = {};
    #pragma unroll 1
    for (int ky = 0; ky < 9; ++ky) {
        #pragma unroll 1
        for (int kx = 0; kx < 9; ++kx) {
            const int kk = ky * 9 + kx;
            const float* ap = abase + ky * 71 + kx;   // + c*5041
            v16bf af, bf;
            #pragma unroll
            for (int e = 0; e < 16; ++e) {
                const int ca = (e & 7) + 8 * LH + ((e >> 3) << 4); // 0..31
                const int cb = (e & 7) + 8 * LH + ((e >> 3) << 4); // 0..31
                af[e] = (__bf16)ap[(size_t)ca * 5041];
                bf[e] = (__bf16)bbase[(size_t)cb * 81 + kk];
            }
            acc = __builtin_amdgcn_wmma_f32_16x16x32_bf16(
                      false, af, false, bf, (short)0, acc, false, false);
        }
    }
    const int rowadd = LH * 8;
    const float bn = b2[n];
    #pragma unroll
    for (int rr = 0; rr < 8; ++rr) {
        const int mm = m0 + rr + rowadd;
        const int bb = mm / 3969, r3 = mm % 3969;
        const int yy = r3 / 63,   xc = r3 % 63;
        out[(((size_t)bb * 16 + n) * 63 + yy) * 63 + xc] = relu_f(acc[rr] + bn);
    }
}

// ---------------------------------------------------------------------------
// Kernel 3: locally-connected conv, one wave per output location.
// D(64x16) = A(64xK) @ B(Kx16), K = KK*16 ordered as k = kk*16 + c so that
// each K=32 step covers 2 spatial taps x 16 channels; element decode is
// pure bit ops.  B (the 251MB unshared weights) loaded once, reused over
// all 4 M-tiles kept in registers.
template <int KSZ, int STRIDE>
__global__ void local_wmma_kernel(const float* __restrict__ in,
                                  const float* __restrict__ wl,
                                  float* __restrict__ out,
                                  int IH, int IW, int OH, int OW) {
    const int lane = threadIdx.x & 31;
    const int LH   = (lane >= 16) ? 1 : 0;
    const int wave = blockIdx.x * (blockDim.x >> 5) + (threadIdx.x >> 5);
    const int nloc = OH * OW;
    if (wave >= nloc) return;                      // wave-uniform
    const int y = wave / OW, x = wave % OW;
    constexpr int KK = KSZ * KSZ;
    const int CH = IH * IW;
    const int n = lane & 15;
    const size_t wc = (size_t)OH * OW * KK;        // stride per c
    const size_t wn = wc * 16;                     // stride per o
    const float* wb = wl + (size_t)n * wn + (size_t)(y * OW + x) * KK; // +c*wc+kk
    const int iy0 = y * STRIDE, ix0 = x * STRIDE;
    const float* ab[4];
    #pragma unroll
    for (int mt = 0; mt < 4; ++mt)
        ab[mt] = in + (size_t)(mt * 16 + (lane & 15)) * 16 * CH;   // +c*CH+d
    v8f acc[4] = {};
    constexpr int KSTEPS = (KK + 1) / 2;
    #pragma unroll 1
    for (int ks = 0; ks < KSTEPS; ++ks) {
        const int kk0 = 2 * ks;
        const int kk1 = kk0 + 1;
        const bool v1 = (kk1 < KK);                // wave-uniform tail guard
        const int ky0 = kk0 / KSZ, kx0 = kk0 % KSZ;
        const int ky1 = v1 ? kk1 / KSZ : 0, kx1 = v1 ? kk1 % KSZ : 0;
        const int d0 = (iy0 + ky0) * IW + ix0 + kx0;
        const int d1 = (iy0 + ky1) * IW + ix0 + kx1;
        v16bf bfrag;
        {
            const int kk = kk0 + LH;               // uniform per half-wave
            const bool ok = (LH == 0) || v1;
            #pragma unroll
            for (int e = 0; e < 16; ++e) {         // B: c = e
                bfrag[e] = (__bf16)(ok ? wb[(size_t)e * wc + kk] : 0.f);
            }
        }
        #pragma unroll
        for (int mt = 0; mt < 4; ++mt) {
            v16bf afrag;
            #pragma unroll
            for (int e = 0; e < 16; ++e) {
                const int c = e;                   // channel
                float v;
                if (LH == 0) v = ab[mt][(size_t)c * CH + d0];            // tap kk0
                else         v = v1 ? ab[mt][(size_t)c * CH + d1] : 0.f; // tap kk0+1
                afrag[e] = (__bf16)v;
            }
            acc[mt] = __builtin_amdgcn_wmma_f32_16x16x32_bf16(
                          false, afrag, false, bfrag, (short)0, acc[mt], false, false);
        }
    }
    const int rowadd = LH * 8;
    #pragma unroll
    for (int mt = 0; mt < 4; ++mt)
        #pragma unroll
        for (int rr = 0; rr < 8; ++rr) {
            const int bb = mt * 16 + rr + rowadd;
            out[(((size_t)bb * 16 + n) * OH + y) * OW + x] = relu_f(acc[mt][rr]);
        }
}

// ---------------------------------------------------------------------------
// Kernel 4: FC layer out(64xN) = A(64xK) @ W(NxK)^T + bias.  Per-lane K
// offsets are contiguous runs -> b128 vector loads for both fragments on
// full steps; wave-uniform scalar-guarded path for the K tail.  W streamed
// exactly once; global_prefetch on the weight stream.
__global__ void fc_wmma_kernel(const float* __restrict__ A,
                               const float* __restrict__ W,
                               const float* __restrict__ bias,
                               float* __restrict__ out,
                               int N, int K, int do_relu) {
    const int lane = threadIdx.x & 31;
    const int LH   = (lane >= 16) ? 1 : 0;
    const int wave = blockIdx.x * (blockDim.x >> 5) + (threadIdx.x >> 5);
    const int ntiles = (N + 15) >> 4;
    if (wave >= ntiles) return;                    // wave-uniform
    const int n0 = wave * 16;
    const int n  = n0 + (lane & 15);
    const int nc = (n < N) ? n : (N - 1);          // clamp OOB weight rows
    const float* wrow = W + (size_t)nc * K;
    const int ksteps = (K + 31) >> 5;
    v8f acc[4] = {};
    #pragma unroll 1
    for (int ks = 0; ks < ksteps; ++ks) {
        const int k0 = ks * 32;
        const bool full = (k0 + 32 <= K);          // wave-uniform
        if (k0 + 96 <= K) __builtin_prefetch(wrow + k0 + 64, 0, 1);
        v16bf bfrag;
        if (full) {
            const float4* bp = (const float4*)(wrow + k0 + 16 * LH);
            #pragma unroll
            for (int q = 0; q < 4; ++q) {
                const float4 t = bp[q];
                bfrag[4 * q + 0] = (__bf16)t.x; bfrag[4 * q + 1] = (__bf16)t.y;
                bfrag[4 * q + 2] = (__bf16)t.z; bfrag[4 * q + 3] = (__bf16)t.w;
            }
        } else {
            #pragma unroll
            for (int e = 0; e < 16; ++e) {
                const int k = k0 + e + 16 * LH;
                bfrag[e] = (__bf16)((k < K) ? wrow[k] : 0.f);
            }
        }
        #pragma unroll
        for (int mt = 0; mt < 4; ++mt) {
            const float* arow = A + (size_t)(mt * 16 + (lane & 15)) * K;
            v16bf afrag;
            if (full) {
                const float4* a0 = (const float4*)(arow + k0 + 16 * LH);
                const float4* a1 = (const float4*)(arow + k0 + 16 * LH + 8);
                const float4 t0 = a0[0], t1 = a0[1], t2 = a1[0], t3 = a1[1];
                afrag[0]  = (__bf16)t0.x; afrag[1]  = (__bf16)t0.y;
                afrag[2]  = (__bf16)t0.z; afrag[3]  = (__bf16)t0.w;
                afrag[4]  = (__bf16)t1.x; afrag[5]  = (__bf16)t1.y;
                afrag[6]  = (__bf16)t1.z; afrag[7]  = (__bf16)t1.w;
                afrag[8]  = (__bf16)t2.x; afrag[9]  = (__bf16)t2.y;
                afrag[10] = (__bf16)t2.z; afrag[11] = (__bf16)t2.w;
                afrag[12] = (__bf16)t3.x; afrag[13] = (__bf16)t3.y;
                afrag[14] = (__bf16)t3.z; afrag[15] = (__bf16)t3.w;
            } else {
                #pragma unroll
                for (int e = 0; e < 16; ++e) {
                    const int k = k0 + e + 16 * LH;
                    afrag[e] = (__bf16)((k < K) ? arow[k] : 0.f);
                }
            }
            acc[mt] = __builtin_amdgcn_wmma_f32_16x16x32_bf16(
                          false, afrag, false, bfrag, (short)0, acc[mt], false, false);
        }
    }
    if (n < N) {
        const int rowadd = LH * 8;
        const float bn = bias[n];
        #pragma unroll
        for (int mt = 0; mt < 4; ++mt)
            #pragma unroll
            for (int rr = 0; rr < 8; ++rr) {
                const int m = mt * 16 + rr + rowadd;
                float v = acc[mt][rr] + bn;
                if (do_relu) v = relu_f(v);
                out[(size_t)m * N + n] = v;
            }
    }
}

// ---------------------------------------------------------------------------
// Kernel 5: row softmax over 1000 classes, one block per batch row.
__global__ void softmax_kernel(const float* __restrict__ in,
                               float* __restrict__ out) {
    __shared__ float red[256];
    const int b = blockIdx.x, t = threadIdx.x;
    const float* row = in + (size_t)b * 1000;
    float mx = -1e30f;
    for (int i = t; i < 1000; i += 256) mx = fmaxf(mx, row[i]);
    red[t] = mx; __syncthreads();
    for (int s = 128; s > 0; s >>= 1) { if (t < s) red[t] = fmaxf(red[t], red[t + s]); __syncthreads(); }
    mx = red[0]; __syncthreads();
    float sum = 0.f;
    for (int i = t; i < 1000; i += 256) sum += expf(row[i] - mx);
    red[t] = sum; __syncthreads();
    for (int s = 128; s > 0; s >>= 1) { if (t < s) red[t] += red[t + s]; __syncthreads(); }
    const float inv = 1.f / red[0];
    for (int i = t; i < 1000; i += 256) out[(size_t)b * 1000 + i] = expf(row[i] - mx) * inv;
}

// ---------------------------------------------------------------------------
extern "C" void kernel_launch(void* const* d_in, const int* in_sizes, int n_in,
                              void* d_out, int out_size, void* d_ws, size_t ws_size,
                              hipStream_t stream) {
    const float* x    = (const float*)d_in[0];   // (64,1,152,152)
    const float* w1   = (const float*)d_in[1];   // (32,1,11,11)
    const float* w2   = (const float*)d_in[2];   // (16,32,9,9)
    const float* b2   = (const float*)d_in[3];   // (16,)
    const float* wl4  = (const float*)d_in[4];   // (16,16,55,55,81)
    const float* wl5  = (const float*)d_in[5];   // (16,16,25,25,49)
    const float* wl6  = (const float*)d_in[6];   // (16,16,21,21,25)
    const float* wfc1 = (const float*)d_in[7];   // (4096,7056)
    const float* bfc1 = (const float*)d_in[8];   // (4096,)
    const float* wfc2 = (const float*)d_in[9];   // (1000,4096)
    const float* bfc2 = (const float*)d_in[10];  // (1000,)
    float* out = (float*)d_out;                  // (64,1000)

    float* ws   = (float*)d_ws;
    float* pool = ws;                            // (64,32,71,71)  10,323,968
    float* c2   = pool + 10323968;               // (64,16,63,63)   4,064,256
    float* l4   = c2   + 4064256;                // (64,16,55,55)   3,097,600
    float* l5   = l4   + 3097600;                // (64,16,25,25)     640,000
    float* l6   = l5   + 640000;                 // (64,16,21,21)     451,584
    float* fc1  = l6   + 451584;                 // (64,4096)         262,144
    float* fc2  = fc1  + 262144;                 // (64,1000)          64,000
    (void)in_sizes; (void)n_in; (void)out_size; (void)ws_size;

    // 1) conv1 + ReLU + pool (fused)
    conv1_fmp_kernel<<<dim3(40, 32, 64), 128, 0, stream>>>(x, w1, pool);

    // 2) conv2 implicit GEMM: 15876 tiles, 8 waves/block
    conv2_wmma_kernel<<<dim3(1985), 256, 0, stream>>>(pool, w2, b2, c2);

    // 3) locally-connected layers (one wave per output location)
    local_wmma_kernel<9, 1><<<dim3((3025 + 7) / 8), 256, 0, stream>>>(c2, wl4, l4, 63, 63, 55, 55);
    local_wmma_kernel<7, 2><<<dim3((625  + 7) / 8), 256, 0, stream>>>(l4, wl5, l5, 55, 55, 25, 25);
    local_wmma_kernel<5, 1><<<dim3((441  + 7) / 8), 256, 0, stream>>>(l5, wl6, l6, 25, 25, 21, 21);

    // 4) fully connected layers
    fc_wmma_kernel<<<dim3(32), 256, 0, stream>>>(l6,  wfc1, bfc1, fc1, 4096, 7056, 1);
    fc_wmma_kernel<<<dim3(8),  256, 0, stream>>>(fc1, wfc2, bfc2, fc2, 1000, 4096, 0);

    // 5) softmax
    softmax_kernel<<<dim3(64), 256, 0, stream>>>(fc2, out);
}
